// SpatialCrossScaleAttention_2568390443354
// MI455X (gfx1250) — compile-verified, matrix operand order checked by reading, not run on hardware
//
#include <hip/hip_runtime.h>

#define DEV_INLINE __device__ __forceinline__

typedef __attribute__((ext_vector_type(16))) __bf16 bf16x16;
typedef __attribute__((ext_vector_type(8)))  float  v8f;

union AFrag {
  bf16x16 v;
  uint4 u[2];
  unsigned short s[16];
};

DEV_INLINE unsigned short f2bf(float f) {
  unsigned int u = __float_as_uint(f);
  u += 0x7FFFu + ((u >> 16) & 1u);   // round-to-nearest-even
  return (unsigned short)(u >> 16);
}

// ---------------------------------------------------------------------------
// f32 -> bf16 convert (weights etc.)
// ---------------------------------------------------------------------------
__global__ void k_f32_to_bf16(const float* __restrict__ in,
                              unsigned short* __restrict__ out, int n) {
  int i = blockIdx.x * blockDim.x + threadIdx.x;
  if (i < n) out[i] = f2bf(in[i]);
}

// ---------------------------------------------------------------------------
// NCHW f32 -> token-major ([B*HW, C]) f32 (optional) + bf16 (optional)
// ---------------------------------------------------------------------------
__global__ void k_nchw_to_tok(const float* __restrict__ in,
                              float* __restrict__ outF,
                              unsigned short* __restrict__ outB,
                              int B, int C, int HW) {
  int i = blockIdx.x * blockDim.x + threadIdx.x;
  int total = B * C * HW;
  if (i >= total) return;
  int c = i % C;
  int t = i / C;            // global token = b*HW + p
  int b = t / HW, p = t % HW;
  float v = in[((size_t)b * C + c) * HW + p];
  if (outF) outF[i] = v;
  if (outB) outB[i] = f2bf(v);
}

// ---------------------------------------------------------------------------
// Bilinear resize (jax.image.resize 'bilinear', antialias=False) from
// token-major f32 src [B*Hs*Ws, 192] into bf16 ctx slice.
// dst element: dst[((size_t)b*Mtok + tokOfs + y*Wd + x)*192 + c]
// ---------------------------------------------------------------------------
__global__ void k_resize_ctx(const float* __restrict__ src, int Hs, int Ws,
                             unsigned short* __restrict__ dst, int Hd, int Wd,
                             int B, int Mtok, int tokOfs) {
  const int C = 192;
  int i = blockIdx.x * blockDim.x + threadIdx.x;
  int total = B * Hd * Wd * C;
  if (i >= total) return;
  int c = i % C;
  int t = i / C;
  int b = t / (Hd * Wd);
  int p = t % (Hd * Wd);
  int y = p / Wd, x = p % Wd;
  float sy = (float)Hs / (float)Hd;
  float sx = (float)Ws / (float)Wd;
  float fy = (y + 0.5f) * sy - 0.5f;
  float fx = (x + 0.5f) * sx - 0.5f;
  int y0 = (int)floorf(fy);
  int x0 = (int)floorf(fx);
  float wy = fy - (float)y0;
  float wx = fx - (float)x0;
  int y0c = min(max(y0, 0), Hs - 1);
  int y1c = min(max(y0 + 1, 0), Hs - 1);
  int x0c = min(max(x0, 0), Ws - 1);
  int x1c = min(max(x0 + 1, 0), Ws - 1);
  const float* base = src + (size_t)b * Hs * Ws * C;
  float v00 = base[((size_t)y0c * Ws + x0c) * C + c];
  float v01 = base[((size_t)y0c * Ws + x1c) * C + c];
  float v10 = base[((size_t)y1c * Ws + x0c) * C + c];
  float v11 = base[((size_t)y1c * Ws + x1c) * C + c];
  float v0 = v00 + (v01 - v00) * wx;
  float v1 = v10 + (v11 - v10) * wx;
  float v = v0 + (v1 - v0) * wy;
  dst[((size_t)b * Mtok + tokOfs + p) * C + c] = f2bf(v);
}

// ---------------------------------------------------------------------------
// Generic WMMA GEMM:  Y[row, col] = X[row, 0:K] . W[col, 0:K] (+bias) (+resid)
// X: bf16 [rows, ldx], W: bf16 [Cout, K] row-major (torch [out,in] layout).
// One wave (32 threads) per 16x16 output tile. grid = (rows/16, Cout/16).
// Optional outputs: bf16 token-major, and f32 token-major (mode 0) or
// channel-major per batch (mode 1, needs rowsPerBatch).
// ---------------------------------------------------------------------------
__global__ void k_gemm_wmma(const unsigned short* __restrict__ X, int ldx,
                            const unsigned short* __restrict__ W,
                            const float* __restrict__ bias,
                            const float* __restrict__ resid,
                            unsigned short* __restrict__ outB,
                            float* __restrict__ outF, int outFMode,
                            int rowsPerBatch, int K, int Cout) {
  int row0 = blockIdx.x * 16;
  int col0 = blockIdx.y * 16;
  int l  = threadIdx.x;
  int lm = l & 15;
  int lo = (l < 16) ? 0 : 8;   // K-interleave offset per lane half

  v8f acc = {};
  const unsigned short* xr = X + (size_t)(row0 + lm) * ldx + lo;
  const unsigned short* wr = W + (size_t)(col0 + lm) * K + lo;
  for (int k0 = 0; k0 < K; k0 += 32) {
    AFrag a, b;
    a.u[0] = *(const uint4*)(xr + k0);
    a.u[1] = *(const uint4*)(xr + k0 + 16);
    b.u[0] = *(const uint4*)(wr + k0);
    b.u[1] = *(const uint4*)(wr + k0 + 16);
    acc = __builtin_amdgcn_wmma_f32_16x16x32_bf16(
        false, a.v, false, b.v, (short)0, acc, false, false);
  }
  int col = col0 + lm;
  float bv = bias ? bias[col] : 0.0f;
  int mb = (l < 16) ? 0 : 8;
#pragma unroll
  for (int r = 0; r < 8; ++r) {
    int row = row0 + mb + r;
    float v = acc[r] + bv;
    if (resid) v += resid[(size_t)row * Cout + col];
    if (outB) outB[(size_t)row * Cout + col] = f2bf(v);
    if (outF) {
      if (outFMode == 0) {
        outF[(size_t)row * Cout + col] = v;
      } else {
        int bb = row / rowsPerBatch;
        int nn = row % rowsPerBatch;
        outF[((size_t)bb * Cout + col) * rowsPerBatch + nn] = v;
      }
    }
  }
}

// ---------------------------------------------------------------------------
// Transpose V out of KV: KV [B*M, 384] (V = cols 192+h*24+d) -> Vt [B,8,24,M]
// ---------------------------------------------------------------------------
__global__ void k_transpose_v(const unsigned short* __restrict__ KV,
                              unsigned short* __restrict__ Vt, int B, int M) {
  int i = blockIdx.x * blockDim.x + threadIdx.x;
  int total = B * 8 * 24 * M;
  if (i >= total) return;
  int m = i % M;
  int t = i / M;
  int d = t % 24; t /= 24;
  int h = t % 8;
  int b = t / 8;
  Vt[i] = KV[((size_t)(b * M + m)) * 384 + 192 + h * 24 + d];
}

// ---------------------------------------------------------------------------
// Flash-style cross attention, one wave per (b, head, 16-query tile).
//   S^T = K_tile x Q  (ctx-major D layout => softmax dir is lane-local and
//   exp'd S^T registers map 1:1 onto the A-fragment of P x V: no transpose)
// Q: bf16 [B*N,192]; KV: bf16 [B*M,384] (K at cols 0..191); Vt: [B,8,24,M]
// O: bf16 [B*N,192]. d=24 zero-padded to K=32. M % 32 == 0, N % 16 == 0.
// ---------------------------------------------------------------------------
__global__ void k_attn(const unsigned short* __restrict__ Q,
                       const unsigned short* __restrict__ KV,
                       const unsigned short* __restrict__ Vt,
                       unsigned short* __restrict__ O,
                       int N, int M) {
  const int h    = blockIdx.y;
  const int tpb  = N / 16;
  const int b    = blockIdx.x / tpb;
  const int q0   = (blockIdx.x % tpb) * 16;
  const int l    = threadIdx.x;
  const int lm   = l & 15;
  const int half = l >> 4;
  const int lo   = half ? 8 : 0;
  const uint4 z4 = make_uint4(0u, 0u, 0u, 0u);
  const float cs = 1.4426950408889634f * rsqrtf(24.0f);  // log2(e)/sqrt(d)

  // Q as B-operand: lane = query column, K = padded head dim.
  AFrag qf;
  const unsigned short* qp = Q + (size_t)(b * N + q0 + lm) * 192 + h * 24 + lo;
  qf.u[0] = *(const uint4*)qp;
  qf.u[1] = half ? z4 : *(const uint4*)(qp + 16);  // d 24..31 pad -> 0

  v8f o0 = {}, o1 = {};
  float mrun = -3.0e38f;
  float srun = 0.0f;

  const unsigned short* kbase = KV + (size_t)(b * M) * 384 + h * 24 + lo;
  const unsigned short* vb0 = Vt + ((size_t)(b * 8 + h) * 24 + lm) * M;
  const unsigned short* vb1 =
      Vt + ((size_t)(b * 8 + h) * 24 + 16 + (lm & 7)) * M;  // valid iff lm<8

  for (int c0 = 0; c0 < M; c0 += 32) {
    // K tiles as A-operand (lane = ctx row).
    AFrag k0f, k1f;
    const unsigned short* kp0 = kbase + (size_t)(c0 + lm) * 384;
    const unsigned short* kp1 = kbase + (size_t)(c0 + 16 + lm) * 384;
    k0f.u[0] = *(const uint4*)kp0;
    k0f.u[1] = half ? z4 : *(const uint4*)(kp0 + 16);
    k1f.u[0] = *(const uint4*)kp1;
    k1f.u[1] = half ? z4 : *(const uint4*)(kp1 + 16);

    v8f s0 = {}, s1 = {};
    s0 = __builtin_amdgcn_wmma_f32_16x16x32_bf16(false, k0f.v, false, qf.v,
                                                 (short)0, s0, false, false);
    s1 = __builtin_amdgcn_wmma_f32_16x16x32_bf16(false, k1f.v, false, qf.v,
                                                 (short)0, s1, false, false);

    // Per-query (per lane-pair) online softmax stats.
    float cmax = -3.0e38f;
#pragma unroll
    for (int r = 0; r < 8; ++r) {
      cmax = fmaxf(cmax, s0[r]);
      cmax = fmaxf(cmax, s1[r]);
    }
    cmax = fmaxf(cmax, __shfl_xor(cmax, 16, 32));
    float mnew  = fmaxf(mrun, cmax);
    float alpha = exp2f((mrun - mnew) * cs);
    srun *= alpha;
#pragma unroll
    for (int r = 0; r < 8; ++r) {              // rescale O accumulators
      float av = __shfl(alpha, r + lo, 32);    // row of o[r] is query r+lo
      o0[r] *= av;
      o1[r] *= av;
    }

    // P = exp(S - m): registers drop straight into A-fragment layout.
    AFrag pf;
    float lsum = 0.0f;
#pragma unroll
    for (int r = 0; r < 8; ++r) {
      float p0 = exp2f((s0[r] - mnew) * cs);
      float p1 = exp2f((s1[r] - mnew) * cs);
      lsum += p0 + p1;
      pf.s[r]     = f2bf(p0);
      pf.s[8 + r] = f2bf(p1);
    }
    srun += lsum;
    mrun = mnew;

    // V chunk as B-operand: lane = d column, contiguous K along Vt rows.
    AFrag v0f, v1f;
    v0f.u[0] = *(const uint4*)(vb0 + c0 + lo);
    v0f.u[1] = *(const uint4*)(vb0 + c0 + lo + 16);
    if (lm < 8) {
      v1f.u[0] = *(const uint4*)(vb1 + c0 + lo);
      v1f.u[1] = *(const uint4*)(vb1 + c0 + lo + 16);
    } else {
      v1f.u[0] = z4;
      v1f.u[1] = z4;
    }
    o0 = __builtin_amdgcn_wmma_f32_16x16x32_bf16(false, pf.v, false, v0f.v,
                                                 (short)0, o0, false, false);
    o1 = __builtin_amdgcn_wmma_f32_16x16x32_bf16(false, pf.v, false, v1f.v,
                                                 (short)0, o1, false, false);
  }

  float stot = srun + __shfl_xor(srun, 16, 32);
#pragma unroll
  for (int r = 0; r < 8; ++r) {
    float sq  = __shfl(stot, r + lo, 32);
    float inv = 1.0f / sq;
    unsigned short* orow = O + (size_t)(b * N + q0 + r + lo) * 192 + h * 24;
    orow[lm] = f2bf(o0[r] * inv);
    if (lm < 8) orow[16 + lm] = f2bf(o1[r] * inv);
  }
}

// ---------------------------------------------------------------------------
// Host orchestration
// ---------------------------------------------------------------------------
extern "C" void kernel_launch(void* const* d_in, const int* in_sizes, int n_in,
                              void* d_out, int out_size, void* d_ws,
                              size_t ws_size, hipStream_t stream) {
  (void)in_sizes; (void)n_in; (void)out_size; (void)ws_size;
  const int B = 4, C = 192;
  const int Nn[3] = {1024, 256, 64};
  const int Mm[3] = {2048, 512, 128};
  const int Hh[3] = {32, 16, 8};
  const int Ww[3] = {32, 16, 8};
  const int Cc[3] = {192, 384, 768};

  const float* f0 = (const float*)d_in[0];
  const float* f1 = (const float*)d_in[1];
  const float* f2 = (const float*)d_in[2];
  const float* proj_w[3] = {nullptr, (const float*)d_in[3], (const float*)d_in[5]};
  const float* proj_b[3] = {nullptr, (const float*)d_in[4], (const float*)d_in[6]};
  const float* q_w[3]  = {(const float*)d_in[7],  (const float*)d_in[11], (const float*)d_in[15]};
  const float* kv_w[3] = {(const float*)d_in[8],  (const float*)d_in[12], (const float*)d_in[16]};
  const float* pr_w[3] = {(const float*)d_in[9],  (const float*)d_in[13], (const float*)d_in[17]};
  const float* pr_b[3] = {(const float*)d_in[10], (const float*)d_in[14], (const float*)d_in[18]};
  const float* out_w[3] = {(const float*)d_in[19], (const float*)d_in[21], (const float*)d_in[23]};
  const float* out_b[3] = {(const float*)d_in[20], (const float*)d_in[22], (const float*)d_in[24]};
  float* outp = (float*)d_out;
  const size_t out_ofs[3] = {0, (size_t)B * 192 * 1024, (size_t)B * 192 * 1024 + (size_t)B * 384 * 256};

  size_t off = 0;
  auto alloc = [&](size_t bytes) -> void* {
    void* p = (char*)d_ws + off;
    off += (bytes + 255) & ~(size_t)255;
    return p;
  };
  auto allocF = [&](size_t n) { return (float*)alloc(n * 4); };
  auto allocH = [&](size_t n) { return (unsigned short*)alloc(n * 2); };

  // f32 token-major projected features (residual + resize source)
  float* Pf[3];
  for (int i = 0; i < 3; ++i) Pf[i] = allocF((size_t)B * Nn[i] * C);
  // bf16 token-major activations
  unsigned short *Xb[3], *Ctx[3], *Qb[3], *KVb[3], *Vt[3], *Ob[3], *Eb[3];
  for (int i = 0; i < 3; ++i) {
    Xb[i]  = allocH((size_t)B * Nn[i] * C);
    Ctx[i] = allocH((size_t)B * Mm[i] * C);
    Qb[i]  = allocH((size_t)B * Nn[i] * C);
    KVb[i] = allocH((size_t)B * Mm[i] * 2 * C);
    Vt[i]  = allocH((size_t)B * 8 * 24 * Mm[i]);
    Ob[i]  = allocH((size_t)B * Nn[i] * C);
    Eb[i]  = allocH((size_t)B * Nn[i] * C);
  }
  unsigned short* T1 = allocH((size_t)B * 256 * 384);  // f1 token-major bf16
  unsigned short* T2 = allocH((size_t)B * 64 * 768);   // f2 token-major bf16
  // bf16 weights
  unsigned short *projWb[3] = {nullptr, allocH(192 * 384), allocH(192 * 768)};
  unsigned short *qWb[3], *kvWb[3], *prWb[3], *outWb[3];
  for (int i = 0; i < 3; ++i) {
    qWb[i]   = allocH(192 * 192);
    kvWb[i]  = allocH(384 * 192);
    prWb[i]  = allocH(192 * 192);
    outWb[i] = allocH((size_t)Cc[i] * 192);
  }

  const int TB = 256;
  auto cvt = [&](const float* src, unsigned short* dst, int n) {
    k_f32_to_bf16<<<(n + TB - 1) / TB, TB, 0, stream>>>(src, dst, n);
  };
  // 1) weight conversion
  cvt(proj_w[1], projWb[1], 192 * 384);
  cvt(proj_w[2], projWb[2], 192 * 768);
  for (int i = 0; i < 3; ++i) {
    cvt(q_w[i], qWb[i], 192 * 192);
    cvt(kv_w[i], kvWb[i], 384 * 192);
    cvt(pr_w[i], prWb[i], 192 * 192);
    cvt(out_w[i], outWb[i], Cc[i] * 192);
  }
  // 2) feature layout conversion
  {
    int n0 = B * 192 * 1024;
    k_nchw_to_tok<<<(n0 + TB - 1) / TB, TB, 0, stream>>>(f0, Pf[0], Xb[0], B, 192, 1024);
    int n1 = B * 384 * 256;
    k_nchw_to_tok<<<(n1 + TB - 1) / TB, TB, 0, stream>>>(f1, nullptr, T1, B, 384, 256);
    int n2 = B * 768 * 64;
    k_nchw_to_tok<<<(n2 + TB - 1) / TB, TB, 0, stream>>>(f2, nullptr, T2, B, 768, 64);
  }
  // 3) input projections (f1,f2 -> 192ch), produce f32 + bf16 token-major
  k_gemm_wmma<<<dim3(B * 256 / 16, 192 / 16), 32, 0, stream>>>(
      T1, 384, projWb[1], proj_b[1], nullptr, Xb[1], Pf[1], 0, 0, 384, 192);
  k_gemm_wmma<<<dim3(B * 64 / 16, 192 / 16), 32, 0, stream>>>(
      T2, 768, projWb[2], proj_b[2], nullptr, Xb[2], Pf[2], 0, 0, 768, 192);
  // 4) contexts: bilinear-resized other scales, concatenated in j order
  for (int i = 0; i < 3; ++i) {
    int ofs = 0;
    for (int j = 0; j < 3; ++j) {
      if (j == i) continue;
      int n = B * Hh[i] * Ww[i] * C;
      k_resize_ctx<<<(n + TB - 1) / TB, TB, 0, stream>>>(
          Pf[j], Hh[j], Ww[j], Ctx[i], Hh[i], Ww[i], B, Mm[i], ofs);
      ofs += Hh[i] * Ww[i];
    }
  }
  // 5) per-scale attention pipeline
  for (int i = 0; i < 3; ++i) {
    int N = Nn[i], M = Mm[i];
    // Q = X @ Wq^T
    k_gemm_wmma<<<dim3(B * N / 16, 192 / 16), 32, 0, stream>>>(
        Xb[i], 192, qWb[i], nullptr, nullptr, Qb[i], nullptr, 0, 0, 192, 192);
    // KV = ctx @ Wkv^T
    k_gemm_wmma<<<dim3(B * M / 16, 384 / 16), 32, 0, stream>>>(
        Ctx[i], 192, kvWb[i], nullptr, nullptr, KVb[i], nullptr, 0, 0, 192, 384);
    // V transpose for contiguous-K B fragments
    {
      int n = B * 8 * 24 * M;
      k_transpose_v<<<(n + TB - 1) / TB, TB, 0, stream>>>(KVb[i], Vt[i], B, M);
    }
    // flash attention
    k_attn<<<dim3(B * N / 16, 8), 32, 0, stream>>>(Qb[i], KVb[i], Vt[i], Ob[i], N, M);
    // out-proj + bias + residual
    k_gemm_wmma<<<dim3(B * N / 16, 192 / 16), 32, 0, stream>>>(
        Ob[i], 192, prWb[i], pr_b[i], Pf[i], Eb[i], nullptr, 0, 0, 192, 192);
    // final 1x1 conv, f32 channel-major straight into d_out
    k_gemm_wmma<<<dim3(B * N / 16, Cc[i] / 16), 32, 0, stream>>>(
        Eb[i], 192, outWb[i], out_b[i], nullptr, nullptr, outp + out_ofs[i], 1,
        N, 192, Cc[i]);
  }
}
